// AttnDecoderModule_58128087384090
// MI455X (gfx1250) — matrix operand order repaired and checked by reading, not compiled, not run on hardware
//
#include <hip/hip_runtime.h>
#include <hip/hip_bf16.h>

#define B_    32
#define TIN_  50
#define TOUT_ 50
#define H_    512
#define V_    32000

typedef __attribute__((ext_vector_type(8)))  float  v8f;
typedef __attribute__((ext_vector_type(16))) __bf16 v16bf;

union BF16Frag {
    unsigned short u[16];
    v16bf v;
};

__device__ __forceinline__ unsigned short f32_to_bf16(float f) {
    unsigned int x = __builtin_bit_cast(unsigned int, f);
    x = (x + 0x7FFFu + ((x >> 16) & 1u)) >> 16;   // round-to-nearest-even
    return (unsigned short)x;
}

// A-fragment (16-bit A 16x32, ISA 7.12.2): lane half/lr fixed, element i ->
//   j=i>>1, p=i&1 ; K = ((j<4)?0:16) + half*8 + (j&3)*2 + p
__device__ __forceinline__ v16bf load_afrag(const unsigned short* __restrict__ Arow,
                                            int kk, int half) {
    BF16Frag a;
#pragma unroll
    for (int i = 0; i < 16; ++i) {
        int j = i >> 1, p = i & 1;
        int ka = ((j < 4) ? 0 : 16) + half * 8 + (j & 3) * 2 + p;
        a.u[i] = Arow[kk + ka];
    }
    return a.v;
}

// B-fragment (32x16): element i -> K = ((i<8)?0:16) + half*8 + (i&7)
__device__ __forceinline__ v16bf load_bfrag(const unsigned short* __restrict__ Brow,
                                            int kk, int half) {
    BF16Frag b;
#pragma unroll
    for (int i = 0; i < 16; ++i)
        b.u[i] = Brow[kk + ((i < 8) ? 0 : 16) + half * 8 + (i & 7)];
    return b.v;
}

// ---------------------------------------------------------------------------
// fp32 -> bf16 bulk convert
// ---------------------------------------------------------------------------
__global__ void k_cvt_bf16(const float* __restrict__ src,
                           unsigned short* __restrict__ dst, int n) {
    int i = blockIdx.x * blockDim.x + threadIdx.x;
    if (i < n) dst[i] = f32_to_bf16(src[i]);
}

// ---------------------------------------------------------------------------
// Register-blocked WMMA bf16 GEMM: each wave computes a 32x32 macro-tile
// (2 M-tiles x 2 N-tiles, 4 accumulators). B fragments are reused across both
// M-tiles -> B-matrix (the large weight) is streamed from L2 exactly once.
// C[M,N] = A[M,K] * Bt[N,K]^T (+bias). Macro-tile count must be a multiple of
// waves/block so EXEC stays all-ones around v_wmma (ISA 7.12 requirement).
// ---------------------------------------------------------------------------
__global__ void k_wmma_gemm_bf16_2x2(const unsigned short* __restrict__ A,
                                     const unsigned short* __restrict__ Bt,
                                     const float* __restrict__ bias,
                                     float* __restrict__ C,
                                     int npairsN, int K, int ldc) {
    int wave = threadIdx.x >> 5;
    int tile = blockIdx.x * (blockDim.x >> 5) + wave;
    int mt   = (tile / npairsN) * 32;
    int nt   = (tile % npairsN) * 32;
    int lane = threadIdx.x & 31;
    int half = lane >> 4;      // 0: lanes 0-15, 1: lanes 16-31
    int lr   = lane & 15;      // row (A) / col (B) index within 16-tile

    const unsigned short* Arow0 = A  + (size_t)(mt + lr) * K;
    const unsigned short* Arow1 = A  + (size_t)(mt + 16 + lr) * K;
    const unsigned short* Brow0 = Bt + (size_t)(nt + lr) * K;
    const unsigned short* Brow1 = Bt + (size_t)(nt + 16 + lr) * K;

    v8f acc00 = {}, acc01 = {}, acc10 = {}, acc11 = {};
    for (int kk = 0; kk < K; kk += 32) {
        __builtin_prefetch(Brow0 + kk + 128, 0, 0);   // global_prefetch_b8
        __builtin_prefetch(Brow1 + kk + 128, 0, 0);
        v16bf a0 = load_afrag(Arow0, kk, half);
        v16bf a1 = load_afrag(Arow1, kk, half);
        v16bf b0 = load_bfrag(Brow0, kk, half);
        v16bf b1 = load_bfrag(Brow1, kk, half);
        acc00 = __builtin_amdgcn_wmma_f32_16x16x32_bf16(
            false, a0, false, b0, (short)0, acc00, false, false);
        acc01 = __builtin_amdgcn_wmma_f32_16x16x32_bf16(
            false, a0, false, b1, (short)0, acc01, false, false);
        acc10 = __builtin_amdgcn_wmma_f32_16x16x32_bf16(
            false, a1, false, b0, (short)0, acc10, false, false);
        acc11 = __builtin_amdgcn_wmma_f32_16x16x32_bf16(
            false, a1, false, b1, (short)0, acc11, false, false);
    }

    int n0 = nt + lr, n1 = nt + 16 + lr;
    float bv0 = bias ? bias[n0] : 0.0f;
    float bv1 = bias ? bias[n1] : 0.0f;
#pragma unroll
    for (int r = 0; r < 8; ++r) {
        int m0 = mt + half * 8 + r;        // C/D layout: VGPR r -> M=r / M=8+r
        int m1 = m0 + 16;
        C[(size_t)m0 * ldc + n0] = acc00[r] + bv0;
        C[(size_t)m0 * ldc + n1] = acc01[r] + bv1;
        C[(size_t)m1 * ldc + n0] = acc10[r] + bv0;
        C[(size_t)m1 * ldc + n1] = acc11[r] + bv1;
    }
}

// ---------------------------------------------------------------------------
// state init: h = 0, tok = y[:,0]
// ---------------------------------------------------------------------------
__global__ void k_init(float* __restrict__ h, unsigned short* __restrict__ hb16,
                       int* __restrict__ tok, const int* __restrict__ y) {
    int gid = blockIdx.x * blockDim.x + threadIdx.x;
    if (gid < B_ * H_) { h[gid] = 0.0f; hb16[gid] = 0; }
    if (gid < B_) tok[gid] = y[gid * TOUT_];
}

// ---------------------------------------------------------------------------
// x = emb[tok];  wh = [x,h] @ W_fc1^T      (one thread per (b,o))
// ---------------------------------------------------------------------------
__global__ void k_wh(const float* __restrict__ emb, const float* __restrict__ W_fc1,
                     const int* __restrict__ tok, const float* __restrict__ h,
                     float* __restrict__ x, float* __restrict__ wh) {
    int gid = blockIdx.x * blockDim.x + threadIdx.x;     // B_*H_
    int b = gid / H_, o = gid % H_;
    const float* xe = emb + (size_t)tok[b] * H_;
    x[gid] = xe[o];
    const float* wr = W_fc1 + (size_t)o * (2 * H_);
    const float* hb = h + (size_t)b * H_;
    float s = 0.0f;
    for (int j = 0; j < H_; ++j) s += xe[j] * wr[j];
    for (int j = 0; j < H_; ++j) s += hb[j] * wr[H_ + j];
    wh[gid] = s;
}

// ---------------------------------------------------------------------------
// attention: logits -> softmax -> context; also writes broadcast attn output
// one block per batch row
// ---------------------------------------------------------------------------
__global__ void k_attn(const float* __restrict__ enc, const float* __restrict__ wenc,
                       const float* __restrict__ W_fc, const float* __restrict__ b_fc,
                       const float* __restrict__ wh, float* __restrict__ attn_x,
                       float* __restrict__ attn_out, int t) {
    int b = blockIdx.x, tid = threadIdx.x;
    __shared__ float lg[TIN_];
    if (tid < TIN_) {
        const float* wrow = wenc + ((size_t)b * TIN_ + tid) * H_;
        const float* whb  = wh + (size_t)b * H_;
        float s = 0.0f;
        for (int hh = 0; hh < H_; ++hh) s += (whb[hh] + wrow[hh]) * W_fc[hh];
        lg[tid] = s + b_fc[0];
    }
    __syncthreads();
    if (tid == 0) {
        float m = lg[0];
        for (int i = 1; i < TIN_; ++i) m = fmaxf(m, lg[i]);
        float sum = 0.0f;
        for (int i = 0; i < TIN_; ++i) { lg[i] = __expf(lg[i] - m); sum += lg[i]; }
        float inv = 1.0f / sum;
        for (int i = 0; i < TIN_; ++i) lg[i] *= inv;
    }
    __syncthreads();
    for (int hh = tid; hh < H_; hh += blockDim.x) {
        float s = 0.0f;
        for (int i = 0; i < TIN_; ++i) s += lg[i] * enc[((size_t)b * TIN_ + i) * H_ + hh];
        attn_x[(size_t)b * H_ + hh] = s;
    }
    float* arow = attn_out + ((size_t)t * B_ + b) * (size_t)(TIN_ * H_);
    for (int idx = tid; idx < TIN_ * H_; idx += blockDim.x)
        arow[idx] = lg[idx / H_];
}

// ---------------------------------------------------------------------------
// xc = relu([x, attn_x] @ W_comb^T + b_comb)
// ---------------------------------------------------------------------------
__global__ void k_comb(const float* __restrict__ x, const float* __restrict__ attn_x,
                       const float* __restrict__ W_comb, const float* __restrict__ b_comb,
                       float* __restrict__ xc) {
    int gid = blockIdx.x * blockDim.x + threadIdx.x;     // B_*H_
    int b = gid / H_, o = gid % H_;
    const float* wr = W_comb + (size_t)o * (2 * H_);
    const float* xb = x + (size_t)b * H_;
    const float* ab = attn_x + (size_t)b * H_;
    float s = b_comb[o];
    for (int j = 0; j < H_; ++j) s += xb[j] * wr[j];
    for (int j = 0; j < H_; ++j) s += ab[j] * wr[H_ + j];
    xc[gid] = fmaxf(s, 0.0f);
}

// ---------------------------------------------------------------------------
// gi = xc @ W_ih^T + b_ih ;  gh = h @ W_hh^T + b_hh
// ---------------------------------------------------------------------------
__global__ void k_gates(const float* __restrict__ xc, const float* __restrict__ h,
                        const float* __restrict__ W_ih, const float* __restrict__ W_hh,
                        const float* __restrict__ b_ih, const float* __restrict__ b_hh,
                        float* __restrict__ gi, float* __restrict__ gh) {
    int gid = blockIdx.x * blockDim.x + threadIdx.x;     // B_*3H_
    int b = gid / (3 * H_), o = gid % (3 * H_);
    const float* xb = xc + (size_t)b * H_;
    const float* hb = h + (size_t)b * H_;
    const float* wi = W_ih + (size_t)o * H_;
    const float* wr = W_hh + (size_t)o * H_;
    float si = b_ih[o], sh = b_hh[o];
    for (int j = 0; j < H_; ++j) { si += xb[j] * wi[j]; sh += hb[j] * wr[j]; }
    gi[gid] = si;
    gh[gid] = sh;
}

// ---------------------------------------------------------------------------
// GRU pointwise update; emits bf16 h for the vocab WMMA GEMM
// ---------------------------------------------------------------------------
__global__ void k_gru(const float* __restrict__ gi, const float* __restrict__ gh,
                      float* __restrict__ h, unsigned short* __restrict__ hb16,
                      float* __restrict__ hT, int last) {
    int gid = blockIdx.x * blockDim.x + threadIdx.x;     // B_*H_
    int b = gid / H_, o = gid % H_;
    const float* gib = gi + (size_t)b * 3 * H_;
    const float* ghb = gh + (size_t)b * 3 * H_;
    float ir = gib[o], iz = gib[H_ + o], inn = gib[2 * H_ + o];
    float hr = ghb[o], hz = ghb[H_ + o], hn  = ghb[2 * H_ + o];
    float r = 1.0f / (1.0f + __expf(-(ir + hr)));
    float z = 1.0f / (1.0f + __expf(-(iz + hz)));
    float n = tanhf(inn + r * hn);
    float hnew = (1.0f - z) * n + z * h[gid];
    h[gid] = hnew;
    hb16[gid] = f32_to_bf16(hnew);
    if (last) hT[gid] = hnew;
}

// ---------------------------------------------------------------------------
// log-softmax over V + argmax + strided write into out[b, v, t]
// one block per batch row
// ---------------------------------------------------------------------------
__global__ void k_logsoftmax(const float* __restrict__ lv, float* __restrict__ out,
                             int* __restrict__ tok, int t) {
    const int NT = 256;
    int b = blockIdx.x, tid = threadIdx.x;
    __shared__ float smax[NT];
    __shared__ int   sidx[NT];
    __shared__ float ssum[NT];
    const float* row = lv + (size_t)b * V_;

    float m = -3.4e38f; int mi = 0;
    for (int v = tid; v < V_; v += NT) {
        float xv = row[v];
        if (xv > m) { m = xv; mi = v; }
    }
    smax[tid] = m; sidx[tid] = mi;
    __syncthreads();
    for (int s = NT / 2; s > 0; s >>= 1) {
        if (tid < s && smax[tid + s] > smax[tid]) {
            smax[tid] = smax[tid + s]; sidx[tid] = sidx[tid + s];
        }
        __syncthreads();
    }
    float rmax = smax[0];

    float sum = 0.0f;
    for (int v = tid; v < V_; v += NT) sum += __expf(row[v] - rmax);
    ssum[tid] = sum;
    __syncthreads();
    for (int s = NT / 2; s > 0; s >>= 1) {
        if (tid < s) ssum[tid] += ssum[tid + s];
        __syncthreads();
    }
    float lse = rmax + __logf(ssum[0]);

    for (int v = tid; v < V_; v += NT)
        out[((size_t)b * V_ + v) * TOUT_ + t] = row[v] - lse;
    if (tid == 0) tok[b] = sidx[0];
}

// ---------------------------------------------------------------------------
extern "C" void kernel_launch(void* const* d_in, const int* in_sizes, int n_in,
                              void* d_out, int out_size, void* d_ws, size_t ws_size,
                              hipStream_t stream) {
    const float* enc    = (const float*)d_in[0];
    const int*   y      = (const int*)d_in[1];
    const float* emb    = (const float*)d_in[2];
    const float* W_fc   = (const float*)d_in[3];
    const float* b_fc   = (const float*)d_in[4];
    const float* W_fc1  = (const float*)d_in[5];
    const float* W_fc2  = (const float*)d_in[6];
    const float* W_comb = (const float*)d_in[7];
    const float* b_comb = (const float*)d_in[8];
    const float* W_ih   = (const float*)d_in[9];
    const float* W_hh   = (const float*)d_in[10];
    const float* b_ih   = (const float*)d_in[11];
    const float* b_hh   = (const float*)d_in[12];
    const float* W_out  = (const float*)d_in[13];
    const float* b_out  = (const float*)d_in[14];

    float* out_logp = (float*)d_out;                           // (B, V, TOUT)
    float* out_hT   = out_logp + (size_t)B_ * V_ * TOUT_;      // (1, B, H)
    float* out_attn = out_hT + (size_t)B_ * H_;                // (TOUT*B, TIN, H)

    // workspace carve-up (256B aligned)
    char* ws = (char*)d_ws;
    size_t off = 0;
    auto carve = [&](size_t bytes) -> char* {
        char* p = ws + off;
        off = (off + bytes + 255) & ~(size_t)255;
        return p;
    };
    float*          wenc   = (float*)carve((size_t)B_ * TIN_ * H_ * 4);
    unsigned short* woutb  = (unsigned short*)carve((size_t)V_ * H_ * 2);
    unsigned short* encb   = (unsigned short*)carve((size_t)B_ * TIN_ * H_ * 2);
    unsigned short* wfc2b  = (unsigned short*)carve((size_t)H_ * H_ * 2);
    float*          hbuf   = (float*)carve((size_t)B_ * H_ * 4);
    unsigned short* hb16   = (unsigned short*)carve((size_t)B_ * H_ * 2);
    float*          xbuf   = (float*)carve((size_t)B_ * H_ * 4);
    float*          whbuf  = (float*)carve((size_t)B_ * H_ * 4);
    float*          axbuf  = (float*)carve((size_t)B_ * H_ * 4);
    float*          xcbuf  = (float*)carve((size_t)B_ * H_ * 4);
    float*          gibuf  = (float*)carve((size_t)B_ * 3 * H_ * 4);
    float*          ghbuf  = (float*)carve((size_t)B_ * 3 * H_ * 4);
    float*          lvbuf  = (float*)carve((size_t)B_ * V_ * 4);
    int*            tokbuf = (int*)carve((size_t)B_ * 4);
    (void)ws_size; (void)in_sizes; (void)n_in; (void)out_size;

    // --- one-time: bf16 conversions ---
    {
        int n = V_ * H_;
        k_cvt_bf16<<<(n + 255) / 256, 256, 0, stream>>>(W_out, woutb, n);
        n = B_ * TIN_ * H_;
        k_cvt_bf16<<<(n + 255) / 256, 256, 0, stream>>>(enc, encb, n);
        n = H_ * H_;
        k_cvt_bf16<<<(n + 255) / 256, 256, 0, stream>>>(W_fc2, wfc2b, n);
    }

    // --- one-time: wenc = enc(1600x512) @ W_fc2(512x512)^T via WMMA bf16 ---
    // 32x32 macro-tiles: (1600/32)*(512/32) = 50*16 = 800 waves -> 200 blocks
    k_wmma_gemm_bf16_2x2<<<200, 128, 0, stream>>>(encb, wfc2b, nullptr, wenc,
                                                  /*npairsN=*/H_ / 32, /*K=*/H_,
                                                  /*ldc=*/H_);

    // --- one-time: state init ---
    k_init<<<(B_ * H_ + 255) / 256, 256, 0, stream>>>(hbuf, hb16, tokbuf, y);

    // --- sequential decode scan ---
    for (int t = 0; t < TOUT_; ++t) {
        k_wh<<<B_ * H_ / 256, 256, 0, stream>>>(emb, W_fc1, tokbuf, hbuf,
                                                xbuf, whbuf);
        k_attn<<<B_, 256, 0, stream>>>(enc, wenc, W_fc, b_fc, whbuf,
                                       axbuf, out_attn, t);
        k_comb<<<B_ * H_ / 256, 256, 0, stream>>>(xbuf, axbuf, W_comb, b_comb,
                                                  xcbuf);
        k_gates<<<B_ * 3 * H_ / 256, 256, 0, stream>>>(xcbuf, hbuf, W_ih, W_hh,
                                                       b_ih, b_hh, gibuf, ghbuf);
        k_gru<<<B_ * H_ / 256, 256, 0, stream>>>(gibuf, ghbuf, hbuf, hb16,
                                                 out_hT, t == TOUT_ - 1);
        // vocab GEMM: h(32x512) @ W_out(32000x512)^T via WMMA bf16
        // 32x32 macro-tiles: (32/32)*(32000/32) = 1000 waves -> 250 blocks
        k_wmma_gemm_bf16_2x2<<<250, 128, 0, stream>>>(hb16, woutb, b_out, lvbuf,
                                                      /*npairsN=*/V_ / 32,
                                                      /*K=*/H_, /*ldc=*/V_);
        k_logsoftmax<<<B_, 256, 0, stream>>>(lvbuf, out_logp, tokbuf, t);
    }
}
